// RNN_3453153705991
// MI455X (gfx1250) — matrix operand order repaired and checked by reading, not compiled, not run on hardware
//
#include <hip/hip_runtime.h>
#include <hip/hip_bf16.h>
#include <math.h>

typedef __attribute__((ext_vector_type(16))) _Float16 v16h;
typedef __attribute__((ext_vector_type(8)))  _Float16 v8h;
typedef __attribute__((ext_vector_type(8)))  float    v8f;

#define BATCH 64
#define SEQ   512
#define IDIM  512
#define HDIM  1024
#define ODIM  512

#define NBLK    32
#define THREADS 256
#define WAVES_PER_BLK (THREADS / 32)

// ---------------- workspace layout (bytes) ----------------
// WxP  : IDIM*HDIM f16 packed B-layout          @ 0         (1 MB)
// WhP  : HDIM*HDIM f16 packed B-layout          @ 1048576   (2 MB)
// WyP  : HDIM*ODIM f16 packed B-layout          @ 3145728   (1 MB)
// biasH: HDIM f32 (Wx_b + Wh_b)                 @ 4194304   (4 KB)
// hA   : BATCH*HDIM f16                         @ 4198400   (128 KB)
// hB   : BATCH*HDIM f16                         @ 4329472   (128 KB)
// bar  : 2 x u32 (count, generation)            @ 4460544
#define WS_WXP 0
#define WS_WHP 1048576
#define WS_WYP 3145728
#define WS_BIASH 4194304
#define WS_HA 4198400
#define WS_HB 4329472
#define WS_BAR 4460544

// Pack a row-major [K][N] f32 weight into WMMA B-operand layout:
// packed[((chunk*(N/16) + nTile)*32 + lane)*16 + h]
//   = W[(chunk*32 + (h<8 ? 8g+h : 16+8g+(h-8))) * N + nTile*16 + (lane&15)], g = lane>>4
// (B 32x16 f16: lane group g holds K=8g..8g+7 and 16+8g..16+8g+7; lane%16 = column)
__global__ void pack_b_kernel(const float* __restrict__ W, _Float16* __restrict__ out,
                              int K, int N) {
    int idx = blockIdx.x * blockDim.x + threadIdx.x;
    if (idx >= K * N) return;
    int h    = idx & 15;
    int lane = (idx >> 4) & 31;
    int rest = idx >> 9;
    int nTiles = N >> 4;
    int nT    = rest % nTiles;
    int chunk = rest / nTiles;
    int k = chunk * 32 + ((h < 8) ? 8 * (lane >> 4) + h : 16 + 8 * (lane >> 4) + (h - 8));
    int n = nT * 16 + (lane & 15);
    out[idx] = (_Float16)W[(size_t)k * N + n];
}

// zero h0, fuse biases, reset barrier
__global__ void init_kernel(const float* __restrict__ Wx_b, const float* __restrict__ Wh_b,
                            float* __restrict__ biasH, _Float16* __restrict__ hA,
                            unsigned* __restrict__ bar) {
    int idx = blockIdx.x * blockDim.x + threadIdx.x;
    if (idx < BATCH * HDIM) hA[idx] = (_Float16)0.0f;
    if (idx < HDIM) biasH[idx] = Wx_b[idx] + Wh_b[idx];
    if (idx < 2) bar[idx] = 0u;
}

__device__ __forceinline__ void grid_barrier(unsigned* cnt, unsigned* gen) {
    __threadfence();          // make this block's stores device-visible
    __syncthreads();
    if (threadIdx.x == 0) {
        unsigned g = __hip_atomic_load(gen, __ATOMIC_RELAXED, __HIP_MEMORY_SCOPE_AGENT);
        unsigned arrived = __hip_atomic_fetch_add(cnt, 1u, __ATOMIC_ACQ_REL,
                                                  __HIP_MEMORY_SCOPE_AGENT);
        if (arrived == NBLK - 1u) {
            __hip_atomic_store(cnt, 0u, __ATOMIC_RELAXED, __HIP_MEMORY_SCOPE_AGENT);
            __hip_atomic_store(gen, g + 1u, __ATOMIC_RELEASE, __HIP_MEMORY_SCOPE_AGENT);
        } else {
            while (__hip_atomic_load(gen, __ATOMIC_ACQUIRE, __HIP_MEMORY_SCOPE_AGENT) == g) {
                __builtin_amdgcn_s_sleep(2);
            }
        }
    }
    __syncthreads();
    __threadfence();          // acquire: don't let stale lines satisfy later loads
}

// Assemble A fragment (16x32 f16) from a contiguous f16 row:
// lane group g: halves 0..7 <- row[kb + 8g .. +7], halves 8..15 <- row[kb+16+8g .. +7]
__device__ __forceinline__ v16h load_a_f16(const _Float16* __restrict__ row, int kb, int grp) {
    const _Float16* p = row + kb + 8 * grp;
    v8h lo = *(const v8h*)(p);
    v8h hi = *(const v8h*)(p + 16);
    return __builtin_shufflevector(lo, hi, 0,1,2,3,4,5,6,7,8,9,10,11,12,13,14,15);
}

__device__ __forceinline__ v16h load_a_f32(const float* __restrict__ row, int kb, int grp) {
    const float* p = row + kb + 8 * grp;
    v16h a;
#pragma unroll
    for (int i = 0; i < 8; ++i) {
        a[i]     = (_Float16)p[i];
        a[8 + i] = (_Float16)p[16 + i];
    }
    return a;
}

__global__ __launch_bounds__(THREADS)
void rnn_persistent_kernel(const float* __restrict__ x,        // [B][T][I]
                           const float* __restrict__ Wy_b,     // [O]
                           const _Float16* __restrict__ WxP,
                           const _Float16* __restrict__ WhP,
                           const _Float16* __restrict__ WyP,
                           const float* __restrict__ biasH,
                           _Float16* __restrict__ hA,
                           _Float16* __restrict__ hB,
                           float* __restrict__ out,            // [B][T][O]
                           unsigned* __restrict__ bar) {
    const int lane = threadIdx.x & 31;
    const int grp  = lane >> 4;       // 0 or 1
    const int ln   = lane & 15;
    const int w    = blockIdx.x * WAVES_PER_BLK + (threadIdx.x >> 5);  // 0..255

    // ---- phase-1 tile (h' = tanh(x@Wx + h@Wh + bias)): 4 x 64 tiles, 1 per wave
    const int mT1   = w >> 6;            // 0..3
    const int nT1   = w & 63;            // 0..63
    const int row1  = mT1 * 16 + ln;     // batch row this lane loads for A
    const int col1  = nT1 * 16 + ln;     // output column in D layout
    const int rb1   = mT1 * 16 + grp * 8;
    const float bh  = biasH[col1];

    // ---- phase-2 tile (y = h'@Wy + by): 4 x 32 tiles, first 128 waves
    const bool doY  = (w < 128);
    const int mT2   = w >> 5;            // 0..3
    const int oT2   = w & 31;            // 0..31
    const int row2  = mT2 * 16 + ln;
    const int col2  = oT2 * 16 + ln;
    const float by  = doY ? Wy_b[col2] : 0.0f;

    unsigned* cnt = bar;
    unsigned* gen = bar + 1;

    for (int t = 0; t < SEQ; ++t) {
        const _Float16* hcur  = (t & 1) ? hB : hA;
        _Float16*       hnext = (t & 1) ? hA : hB;

        // ---------------- phase 1: hidden update ----------------
        v8f acc;
#pragma unroll
        for (int r = 0; r < 8; ++r) acc[r] = bh;

        // x contribution: K = IDIM = 512 -> 16 chunks of 32
        const float* xrow = x + ((size_t)row1 * SEQ + t) * IDIM;
#pragma unroll 2
        for (int c = 0; c < IDIM / 32; ++c) {
            v16h a = load_a_f32(xrow, c * 32, grp);
            v16h b = *(const v16h*)(WxP + (((size_t)c * (HDIM / 16) + nT1) * 32 + lane) * 16);
            acc = __builtin_amdgcn_wmma_f32_16x16x32_f16(false, a, false, b,
                                                         (short)0, acc, false, false);
        }
        // h contribution: K = HDIM = 1024 -> 32 chunks of 32
        const _Float16* hrow = hcur + (size_t)row1 * HDIM;
#pragma unroll 2
        for (int c = 0; c < HDIM / 32; ++c) {
            v16h a = load_a_f16(hrow, c * 32, grp);
            v16h b = *(const v16h*)(WhP + (((size_t)c * (HDIM / 16) + nT1) * 32 + lane) * 16);
            acc = __builtin_amdgcn_wmma_f32_16x16x32_f16(false, a, false, b,
                                                         (short)0, acc, false, false);
        }
        // tanh + scatter-store h' (f16, row-major [b][n])
        {
            _Float16* hw = hnext + (size_t)rb1 * HDIM + col1;
#pragma unroll
            for (int r = 0; r < 8; ++r) hw[(size_t)r * HDIM] = (_Float16)tanhf(acc[r]);
        }

        grid_barrier(cnt, gen);   // h'(t) fully visible device-wide

        // ---------------- phase 2: output projection ----------------
        if (doY) {
            v8f accy;
#pragma unroll
            for (int r = 0; r < 8; ++r) accy[r] = by;
            const _Float16* hrow2 = hnext + (size_t)row2 * HDIM;
#pragma unroll 2
            for (int c = 0; c < HDIM / 32; ++c) {
                v16h a = load_a_f16(hrow2, c * 32, grp);
                v16h b = *(const v16h*)(WyP + (((size_t)c * (ODIM / 16) + oT2) * 32 + lane) * 16);
                accy = __builtin_amdgcn_wmma_f32_16x16x32_f16(false, a, false, b,
                                                              (short)0, accy, false, false);
            }
            const int b0 = mT2 * 16 + grp * 8;
#pragma unroll
            for (int r = 0; r < 8; ++r) {
                out[((size_t)(b0 + r) * SEQ + t) * ODIM + col2] = accy[r];
            }
        }
        // No second barrier needed: next step's writes target the other h buffer,
        // and every cross-step hazard is separated by the next phase-1 barrier.
    }
}

extern "C" void kernel_launch(void* const* d_in, const int* in_sizes, int n_in,
                              void* d_out, int out_size, void* d_ws, size_t ws_size,
                              hipStream_t stream) {
    const float* x    = (const float*)d_in[0];
    const float* Wx_w = (const float*)d_in[1];
    const float* Wx_b = (const float*)d_in[2];
    const float* Wh_w = (const float*)d_in[3];
    const float* Wh_b = (const float*)d_in[4];
    const float* Wy_w = (const float*)d_in[5];
    const float* Wy_b = (const float*)d_in[6];
    float* out = (float*)d_out;

    char* ws = (char*)d_ws;
    _Float16* WxP   = (_Float16*)(ws + WS_WXP);
    _Float16* WhP   = (_Float16*)(ws + WS_WHP);
    _Float16* WyP   = (_Float16*)(ws + WS_WYP);
    float*    biasH = (float*)(ws + WS_BIASH);
    _Float16* hA    = (_Float16*)(ws + WS_HA);
    _Float16* hB    = (_Float16*)(ws + WS_HB);
    unsigned* bar   = (unsigned*)(ws + WS_BAR);

    // pack weights into WMMA B-operand layout (f16)
    {
        int n = IDIM * HDIM;
        pack_b_kernel<<<(n + 255) / 256, 256, 0, stream>>>(Wx_w, WxP, IDIM, HDIM);
    }
    {
        int n = HDIM * HDIM;
        pack_b_kernel<<<(n + 255) / 256, 256, 0, stream>>>(Wh_w, WhP, HDIM, HDIM);
    }
    {
        int n = HDIM * ODIM;
        pack_b_kernel<<<(n + 255) / 256, 256, 0, stream>>>(Wy_w, WyP, HDIM, ODIM);
    }
    // zero h0, fuse biases, reset barrier (runs every launch -> deterministic)
    {
        int n = BATCH * HDIM;
        init_kernel<<<(n + 255) / 256, 256, 0, stream>>>(Wx_b, Wh_b, biasH, hA, bar);
    }
    // persistent RNN scan: 32 blocks x 8 waves, one device barrier per timestep
    rnn_persistent_kernel<<<NBLK, THREADS, 0, stream>>>(
        x, Wy_b, WxP, WhP, WyP, biasH, hA, hB, out, bar);
}
